// Hierarchical_4028679324227
// MI455X (gfx1250) — matrix operand order repaired and checked by reading, not compiled, not run on hardware
//
#include <hip/hip_runtime.h>
#include <math.h>

// ---------------------------------------------------------------------------
// CDNA5 (gfx1250) WMMA types
// ---------------------------------------------------------------------------
typedef __attribute__((ext_vector_type(16))) __bf16   v16bf;
typedef __attribute__((ext_vector_type(8)))  float    v8f;
typedef __attribute__((ext_vector_type(4)))  unsigned uint4v;   // 16B, 16B-aligned

#define BN 4
#define IMGH 224
#define IMGW 224

__device__ __forceinline__ __bf16 f2bf(float f) {
  unsigned u = __builtin_bit_cast(unsigned, f);
  u = u + 0x7FFFu + ((u >> 16) & 1u);               // round-to-nearest-even
  unsigned short h = (unsigned short)(u >> 16);
  return __builtin_bit_cast(__bf16, h);
}
__device__ __forceinline__ float bf2f(__bf16 h) {
  unsigned short s = __builtin_bit_cast(unsigned short, h);
  unsigned u = ((unsigned)s) << 16;
  return __builtin_bit_cast(float, u);
}

__device__ __forceinline__ float wave_reduce_add(float v) {
  #pragma unroll
  for (int off = 16; off > 0; off >>= 1) v += __shfl_xor(v, off, 32);
  return v;
}

// fp32 -> bf16 bulk convert (weights / input image, once per launch)
__global__ void cvt_bf16_kernel(const float* __restrict__ src,
                                __bf16* __restrict__ dst, int n)
{
  int i = blockIdx.x * blockDim.x + threadIdx.x;
  if (i < n) dst[i] = f2bf(src[i]);
}

// ---------------------------------------------------------------------------
// Implicit-GEMM convolution via V_WMMA_F32_16X16X32_BF16, bf16-resident.
//   GEMM view: M = OC, N = Bn*OH*OW, K = IC*KH*KW (all compile time).
// Block = 128 threads (4 waves) -> 128(M) x 64(N) output tile.
//   - im2col B-tile (32K x 64N bf16) staged once per block into LDS,
//     row stride 40 bf16 (80 B): rows 16B-aligned -> b128 LDS traffic,
//     20n mod 64 banks distinct for n=0..15 -> conflict-free.
//   - per wave: 2 A-fragments (packed bf16 weight rows; 2x b128 global loads
//     each when K%32==0) x 4 B-fragments (2x ds_load_b128 each)
//     = 8 WMMAs / K-step, 8 v8f accumulators (32x64 per wave).
// fp32 accumulate; bias+ReLU fused; store bf16 activations.
// ---------------------------------------------------------------------------
template <int IC, int IH, int IW, int OC, int OH, int OW,
          int KH, int KW, int S, int PH, int PW>
__global__ __launch_bounds__(128)
void conv_wmma(const __bf16* __restrict__ in, const __bf16* __restrict__ wgt,
               const float* __restrict__ bias, __bf16* __restrict__ out, int Bn)
{
  constexpr int  K    = IC * KH * KW;
  constexpr bool KAL  = (K % 32) == 0;             // aligned K: no guards
  constexpr int  BSTR = 40;                        // LDS row stride (bf16)
  __shared__ __bf16 ldsB[64 * BSTR];

  const int tid  = threadIdx.x;
  const int lane = tid & 31;
  const int wave = tid >> 5;
  const int lmod = lane & 15;
  const int lhi  = lane >> 4;                      // 0 or 1

  const int N     = Bn * OH * OW;
  const int mWave = blockIdx.x * 128 + wave * 32;  // 32 M rows per wave
  const int nBase = blockIdx.y * 64;

  const __bf16 zbf = __builtin_bit_cast(__bf16, (unsigned short)0);

  // --- LDS-fill coordinates (fixed across the K loop) --------------------
  const int  nFill   = nBase + (tid & 63);
  const int  kHalf   = (tid >> 6) * 16;            // 0 or 16
  const bool nFillOk = (nFill < N);
  int fb = 0, foh = 0, fow = 0;
  if (nFillOk) {
    fb = nFill / (OH * OW);                        // compile-time divisors
    int r = nFill % (OH * OW);
    foh = r / OW; fow = r % OW;
  }

  // --- A rows for this wave's two fragments ------------------------------
  const __bf16* wrow0 = wgt + (size_t)(mWave + lmod) * K;
  const __bf16* wrow1 = wgt + (size_t)(mWave + 16 + lmod) * K;

  v8f acc[2][4] = {};

  for (int kb = 0; kb < K; kb += 32) {
    __syncthreads();                               // prev-iter reads done
    {
      union { uint4v q[2]; __bf16 h[16]; } fill;
      #pragma unroll
      for (int i = 0; i < 16; ++i) {
        int k = kb + kHalf + i;
        __bf16 val = zbf;
        if (nFillOk && (KAL || k < K)) {
          int ic = k / (KH * KW);
          int rr = k % (KH * KW);
          int kh = rr / KW, kw = rr % KW;
          int ih = foh * S - PH + kh;
          int iw = fow * S - PW + kw;
          if (ih >= 0 && ih < IH && iw >= 0 && iw < IW)
            val = in[(((size_t)fb * IC + ic) * IH + ih) * IW + iw];
        }
        fill.h[i] = val;
      }
      uint4v* dst = (uint4v*)&ldsB[(tid & 63) * BSTR + kHalf];
      dst[0] = fill.q[0];
      dst[1] = fill.q[1];
    }
    __syncthreads();

    // A fragments
    v16bf a0, a1;
    const int klo = kb + (lhi ? 16 : 0);
    const int khi = kb + (lhi ? 24 : 8);
    if constexpr (KAL) {
      union { uint4v q[2]; v16bf v; } ua;
      ua.q[0] = *(const uint4v*)(wrow0 + klo);
      ua.q[1] = *(const uint4v*)(wrow0 + khi);
      a0 = ua.v;
      ua.q[0] = *(const uint4v*)(wrow1 + klo);
      ua.q[1] = *(const uint4v*)(wrow1 + khi);
      a1 = ua.v;
    } else {
      #pragma unroll
      for (int j = 0; j < 4; ++j) {
        int k0 = klo + 2 * j, k2 = khi + 2 * j;
        a0[2 * j]     = (k0     < K) ? wrow0[k0]     : zbf;
        a0[2 * j + 1] = (k0 + 1 < K) ? wrow0[k0 + 1] : zbf;
        a0[8 + 2 * j] = (k2     < K) ? wrow0[k2]     : zbf;
        a0[9 + 2 * j] = (k2 + 1 < K) ? wrow0[k2 + 1] : zbf;
        a1[2 * j]     = (k0     < K) ? wrow1[k0]     : zbf;
        a1[2 * j + 1] = (k0 + 1 < K) ? wrow1[k0 + 1] : zbf;
        a1[8 + 2 * j] = (k2     < K) ? wrow1[k2]     : zbf;
        a1[9 + 2 * j] = (k2 + 1 < K) ? wrow1[k2 + 1] : zbf;
      }
    }

    // 4 B fragments from LDS, 8 WMMAs
    #pragma unroll
    for (int ns = 0; ns < 4; ++ns) {
      union { uint4v q[2]; v16bf v; } ub;
      const uint4v* p =
          (const uint4v*)&ldsB[(ns * 16 + lmod) * BSTR + (lhi ? 16 : 0)];
      ub.q[0] = p[0];
      ub.q[1] = p[1];
      acc[0][ns] = __builtin_amdgcn_wmma_f32_16x16x32_bf16(
          false, a0, false, ub.v, (short)0, acc[0][ns], false, false);
      acc[1][ns] = __builtin_amdgcn_wmma_f32_16x16x32_bf16(
          false, a1, false, ub.v, (short)0, acc[1][ns], false, false);
    }
  }

  // Store: bias + ReLU fused, bf16 activations out
  #pragma unroll
  for (int mf = 0; mf < 2; ++mf) {
    #pragma unroll
    for (int ns = 0; ns < 4; ++ns) {
      int n = nBase + ns * 16 + lmod;
      if (n < N) {
        int bimg = n / (OH * OW);
        int rem  = n % (OH * OW);
        #pragma unroll
        for (int r = 0; r < 8; ++r) {
          int mo = mWave + mf * 16 + r + (lhi ? 8 : 0);
          float v = fmaxf(acc[mf][ns][r] + bias[mo], 0.0f);
          out[((size_t)bimg * OC + mo) * (OH * OW) + rem] = f2bf(v);
        }
      }
    }
  }
}

// ---------------------------------------------------------------------------
// y[b,o] = dot(x[b,:], w[o,:]) + bias[o]    (one wave per output, fp32)
// ---------------------------------------------------------------------------
__global__ void linear_kernel(const float* __restrict__ x, const float* __restrict__ w,
                              const float* __restrict__ bias, float* __restrict__ y,
                              int Bn, int In, int On)
{
  int wid  = (int)((blockIdx.x * blockDim.x + threadIdx.x) >> 5);
  int lane = threadIdx.x & 31;
  if (wid >= Bn * On) return;
  int b = wid / On, o = wid % On;
  const float* xr = x + (size_t)b * In;
  const float* wr = w + (size_t)o * In;
  float s = 0.0f;
  for (int i = lane; i < In; i += 32) s += xr[i] * wr[i];
  s = wave_reduce_add(s);
  if (lane == 0) y[wid] = s + bias[o];
}

__global__ void argmax_kernel(const float* __restrict__ gf, int* __restrict__ top,
                              int Bn, int On)
{
  int b = blockIdx.x * blockDim.x + threadIdx.x;
  if (b >= Bn) return;
  float best = gf[(size_t)b * On];
  int bi = 0;
  for (int o = 1; o < On; ++o) {
    float v = gf[(size_t)b * On + o];
    if (v > best) { best = v; bi = o; }
  }
  top[b] = bi;
}

// cam[b,p] = sigmoid(dot(fmap[b,:,p], wc[top[b],:]) + bc[top[b]])  (wave/out)
__global__ void cam_kernel(const __bf16* __restrict__ fmap, const float* __restrict__ wc,
                           const float* __restrict__ bc, const int* __restrict__ top,
                           float* __restrict__ cam, int Bn, int Cch, int P)
{
  int wid  = (int)((blockIdx.x * blockDim.x + threadIdx.x) >> 5);
  int lane = threadIdx.x & 31;
  if (wid >= Bn * P) return;
  int b = wid / P, p = wid % P;
  int t = top[b];
  const float* wr = wc + (size_t)t * Cch;
  const __bf16* f = fmap + (size_t)b * Cch * P + p;
  float s = 0.0f;
  for (int c = lane; c < Cch; c += 32) s += bf2f(f[(size_t)c * P]) * wr[c];
  s = wave_reduce_add(s);
  if (lane == 0) cam[wid] = 1.0f / (1.0f + expf(-(s + bc[t])));
}

// wsc[b,x] = max_y bilinear_up(cam)(y,x);  hsc[b,y] = max_x ... (align_corners)
__global__ void score_kernel(const float* __restrict__ cam,
                             float* __restrict__ wsc, float* __restrict__ hsc,
                             int Bn, int fh, int fw, int Hh, int Ww)
{
  int t = blockIdx.x * blockDim.x + threadIdx.x;
  if (t >= Bn * (Ww + Hh)) return;
  int b = t / (Ww + Hh);
  int r = t - b * (Ww + Hh);
  const float* c = cam + (size_t)b * fh * fw;
  if (r < Ww) {
    int x = r;
    float xs = x * (float)(fw - 1) / (float)(Ww - 1);
    int xlo = (int)floorf(xs); if (xlo < 0) xlo = 0; if (xlo > fw - 1) xlo = fw - 1;
    int xhi = xlo + 1; if (xhi > fw - 1) xhi = fw - 1;
    float xf = xs - xlo;
    float col[16];
    for (int rr = 0; rr < fh; ++rr)
      col[rr] = c[rr * fw + xlo] * (1.0f - xf) + c[rr * fw + xhi] * xf;
    float m = -1e30f;
    for (int y = 0; y < Hh; ++y) {
      float ys = y * (float)(fh - 1) / (float)(Hh - 1);
      int ylo = (int)floorf(ys); if (ylo < 0) ylo = 0; if (ylo > fh - 1) ylo = fh - 1;
      int yhi = ylo + 1; if (yhi > fh - 1) yhi = fh - 1;
      float yf = ys - ylo;
      m = fmaxf(m, col[ylo] * (1.0f - yf) + col[yhi] * yf);
    }
    wsc[(size_t)b * Ww + x] = m;
  } else {
    int y = r - Ww;
    float ys = y * (float)(fh - 1) / (float)(Hh - 1);
    int ylo = (int)floorf(ys); if (ylo < 0) ylo = 0; if (ylo > fh - 1) ylo = fh - 1;
    int yhi = ylo + 1; if (yhi > fh - 1) yhi = fh - 1;
    float yf = ys - ylo;
    float row[16];
    for (int cc = 0; cc < fw; ++cc)
      row[cc] = c[ylo * fw + cc] * (1.0f - yf) + c[yhi * fw + cc] * yf;
    float m = -1e30f;
    for (int x = 0; x < Ww; ++x) {
      float xs = x * (float)(fw - 1) / (float)(Ww - 1);
      int xlo = (int)floorf(xs); if (xlo < 0) xlo = 0; if (xlo > fw - 1) xlo = fw - 1;
      int xhi = xlo + 1; if (xhi > fw - 1) xhi = fw - 1;
      float xf = xs - xlo;
      m = fmaxf(m, row[xlo] * (1.0f - xf) + row[xhi] * xf);
    }
    hsc[(size_t)b * Hh + y] = m;
  }
}

// norm01 + obj_loc interval search (faithful to the jax reference), n = 224.
// locs[b*4 + 0,1] = x1,x2 (from wsc);  locs[b*4 + 2,3] = y1,y2 (from hsc)
__global__ void objloc_kernel(const float* __restrict__ wsc, const float* __restrict__ hsc,
                              int* __restrict__ locs, int Bn, int n, float thresh)
{
  int t = blockIdx.x * blockDim.x + threadIdx.x;
  if (t >= Bn * 2) return;
  int b = t >> 1, axis = t & 1;
  const float* src = (axis == 0) ? (wsc + (size_t)b * n) : (hsc + (size_t)b * n);

  float s[IMGW];
  float mn = 1e30f, mx = -1e30f;
  for (int i = 0; i < n; ++i) { float v = src[i]; mn = fminf(mn, v); mx = fmaxf(mx, v); }
  for (int i = 0; i < n; ++i) {
    float num = (mx == mn) ? src[i] : (src[i] - mn);
    float den = (mx == mn) ? mx : (mx - mn);
    s[i] = num / den;
  }
  int seg[IMGW];
  int run = 0;
  for (int i = 0; i < n; ++i) {
    int ind = 0;
    if (i < n - 1) {
      int s0 = (s[i]     > thresh) - (s[i]     < thresh);
      int s1 = (s[i + 1] > thresh) - (s[i + 1] < thresh);
      int d = s1 - s0; if (d < 0) d = -d;
      ind = (d == 2) ? 1 : 0;
    }
    run += ind;
    seg[i] = run;                       // inclusive cumsum
  }
  int k = run;

  float smax[IMGW]; int slen[IMGW], sstart[IMGW], send[IMGW];
  for (int i = 0; i <= k; ++i) { smax[i] = -1e30f; slen[i] = 0; sstart[i] = n; send[i] = -1; }
  for (int i = 0; i < n; ++i) {
    int id = seg[i];
    smax[id] = fmaxf(smax[id], s[i]);
    slen[id] += 1;
    if (i < sstart[id]) sstart[id] = i;
    if (i > send[id])   send[id]   = i;
  }
  float gmax = -1e30f;
  for (int i = 0; i <= k; ++i) if (slen[i] > 0) gmax = fmaxf(gmax, smax[i]);
  int best_len = -1;
  for (int i = 0; i <= k; ++i)
    if (slen[i] > 0 && smax[i] == gmax && slen[i] > best_len) best_len = slen[i];
  int sel = 0, best_order = n + 2;
  for (int i = 0; i <= k; ++i) {
    if (slen[i] > 0 && smax[i] == gmax && slen[i] == best_len) {
      int order = (i == 0) ? 0 : ((i == k) ? 1 : (i + 1));
      if (order < best_order) { best_order = order; sel = i; }
    }
  }
  int zmin = sstart[sel], zmax = send[sel] + 1;

  int minsize = (n + 7) / 8;            // ceil(n*0.125)
  int defmax  = (n * 7 + 7) / 8;        // ceil(n*0.875)
  int pad = minsize - (zmax - zmin);
  int hp  = (pad + 1) / 2;
  bool need = (zmax - zmin) <= minsize;
  if (need && (zmin > hp) && ((n - zmax) > pad)) { zmin = zmin - hp + 1; zmax = zmax + hp; }
  if (need && (zmin < hp))                       { zmin = 0; zmax = minsize; }
  if (need && ((n - zmax) < hp))                 { zmin = n - minsize + 1; zmax = n; }
  if (k == 0)                                    { zmin = minsize; zmax = defmax; }

  locs[b * 4 + axis * 2 + 0] = zmin;
  locs[b * 4 + axis * 2 + 1] = zmax;
}

// bilinear (align_corners) crop-resize of img[:, y1:y2, x1:x2] -> bf16 (C,224,224)
__global__ void resize_kernel(const float* __restrict__ img, const int* __restrict__ locs,
                              __bf16* __restrict__ out, int Bn, int C, int Hh, int Ww)
{
  int idx = blockIdx.x * blockDim.x + threadIdx.x;
  int total = Bn * C * Hh * Ww;
  if (idx >= total) return;
  int x = idx % Ww; int t2 = idx / Ww;
  int y = t2 % Hh;  t2 /= Hh;
  int c = t2 % C;   int b = t2 / C;
  int x1 = locs[b * 4 + 0], x2 = locs[b * 4 + 1];
  int y1 = locs[b * 4 + 2], y2 = locs[b * 4 + 3];
  float ys = (float)y1 + y * (float)(y2 - 1 - y1) / (float)(Hh - 1);
  float xs = (float)x1 + x * (float)(x2 - 1 - x1) / (float)(Ww - 1);
  int ylo = (int)floorf(ys); if (ylo < 0) ylo = 0; if (ylo > Hh - 1) ylo = Hh - 1;
  int yhi = ylo + 1; if (yhi > Hh - 1) yhi = Hh - 1;
  float yf = ys - ylo;
  int xlo = (int)floorf(xs); if (xlo < 0) xlo = 0; if (xlo > Ww - 1) xlo = Ww - 1;
  int xhi = xlo + 1; if (xhi > Ww - 1) xhi = Ww - 1;
  float xf = xs - xlo;
  const float* p = img + ((size_t)b * C + c) * Hh * Ww;
  float r0 = p[ylo * Ww + xlo] * (1.0f - xf) + p[ylo * Ww + xhi] * xf;
  float r1 = p[yhi * Ww + xlo] * (1.0f - xf) + p[yhi * Ww + xhi] * xf;
  out[idx] = f2bf(r0 * (1.0f - yf) + r1 * yf);
}

__global__ void avgpool_kernel(const __bf16* __restrict__ in, float* __restrict__ out,
                               int BnC, int P)
{
  int t = blockIdx.x * blockDim.x + threadIdx.x;
  if (t >= BnC) return;
  const __bf16* p = in + (size_t)t * P;
  float s = 0.0f;
  for (int i = 0; i < P; ++i) s += bf2f(p[i]);
  out[t] = s / (float)P;
}

// out[b,o] = (gf + dot(lf,w1[o]) + b1[o] (+ dot(lfprev,w1[o]) + b1[o])) / 2
__global__ void final_kernel(const float* __restrict__ gf, const float* __restrict__ lf,
                             const float* __restrict__ lfprev,
                             const float* __restrict__ w1, const float* __restrict__ b1,
                             float* __restrict__ out, int Bn, int In, int On, int hasprev)
{
  int wid  = (int)((blockIdx.x * blockDim.x + threadIdx.x) >> 5);
  int lane = threadIdx.x & 31;
  if (wid >= Bn * On) return;
  int b = wid / On, o = wid % On;
  const float* wr = w1 + (size_t)o * In;
  const float* xr = lf + (size_t)b * In;
  float s0 = 0.0f, s1 = 0.0f;
  for (int i = lane; i < In; i += 32) {
    float wv = wr[i];
    s0 += xr[i] * wv;
    if (hasprev) s1 += lfprev[(size_t)b * In + i] * wv;
  }
  s0 = wave_reduce_add(s0);
  s1 = wave_reduce_add(s1);
  if (lane == 0) {
    float ls = s0 + b1[o];
    if (hasprev) ls += s1 + b1[o];     // bias added twice, as in reference
    out[wid] = (gf[wid] + ls) * 0.5f;
  }
}

// ---------------------------------------------------------------------------
// Host orchestration
// ---------------------------------------------------------------------------
static inline int cdiv(int a, int b) { return (a + b - 1) / b; }

struct BBParams {
  const __bf16 *w1, *w2, *w3, *w4;     // pre-converted bf16 weights
  const float  *b1, *b2, *b3, *b4;     // fp32 biases
};

static void run_backbone(const __bf16* in, const BBParams& p,
                         __bf16* f1, __bf16* f2, __bf16* f3, __bf16* f4,
                         hipStream_t stream)
{
  dim3 blk(128);
  // conv1: 3 -> 256, 7x7 s4, SAME pad_lo = 1, out 56x56   (K = 147)
  conv_wmma<3, 224, 224, 256, 56, 56, 7, 7, 4, 1, 1>
      <<<dim3(256 / 128, cdiv(BN * 56 * 56, 64)), blk, 0, stream>>>(
          in, p.w1, p.b1, f1, BN);
  // conv2: 256 -> 512, 3x3 s2, SAME pad_lo = 0, out 28x28 (K = 2304)
  conv_wmma<256, 56, 56, 512, 28, 28, 3, 3, 2, 0, 0>
      <<<dim3(512 / 128, cdiv(BN * 28 * 28, 64)), blk, 0, stream>>>(
          f1, p.w2, p.b2, f2, BN);
  // conv3: 512 -> 1024, 1x1 s2, out 14x14                 (K = 512)
  conv_wmma<512, 28, 28, 1024, 14, 14, 1, 1, 2, 0, 0>
      <<<dim3(1024 / 128, cdiv(BN * 14 * 14, 64)), blk, 0, stream>>>(
          f2, p.w3, p.b3, f3, BN);
  // conv4: 1024 -> 2048, 1x1 s2, out 7x7                  (K = 1024)
  conv_wmma<1024, 14, 14, 2048, 7, 7, 1, 1, 2, 0, 0>
      <<<dim3(2048 / 128, cdiv(BN * 7 * 7, 64)), blk, 0, stream>>>(
          f3, p.w4, p.b4, f4, BN);
}

extern "C" void kernel_launch(void* const* d_in, const int* in_sizes, int n_in,
                              void* d_out, int out_size, void* d_ws, size_t ws_size,
                              hipStream_t stream) {
  (void)in_sizes; (void)n_in; (void)out_size; (void)ws_size;
  const float* inputs = (const float*)d_in[0];
  const float* bw1 = (const float*)d_in[1];
  const float* bb1 = (const float*)d_in[2];
  const float* bw2 = (const float*)d_in[3];
  const float* bb2 = (const float*)d_in[4];
  const float* bw3 = (const float*)d_in[5];
  const float* bb3 = (const float*)d_in[6];
  const float* bw4 = (const float*)d_in[7];
  const float* bb4 = (const float*)d_in[8];
  const float* w_ord  = (const float*)d_in[9];
  const float* b_ord  = (const float*)d_in[10];
  const float* w_ord1 = (const float*)d_in[11];
  const float* b_ord1 = (const float*)d_in[12];
  const float* w_fam  = (const float*)d_in[13];
  const float* b_fam  = (const float*)d_in[14];
  const float* w_fam1 = (const float*)d_in[15];
  const float* b_fam1 = (const float*)d_in[16];
  const float* w_sp   = (const float*)d_in[17];
  const float* b_sp   = (const float*)d_in[18];
  const float* w_sp1  = (const float*)d_in[19];
  const float* b_sp1  = (const float*)d_in[20];

  // workspace layout (256B-aligned slots)
  char* base = (char*)d_ws;
  size_t off = 0;
  auto allocB = [&](size_t bytes) {
    void* p = base + off;
    off += (bytes + 255) & ~(size_t)255;
    return p;
  };
  const size_t NIMG = (size_t)BN * 3 * IMGH * IMGW;       //   602,112
  __bf16* inbf = (__bf16*)allocB(NIMG * 2);
  __bf16* wb1  = (__bf16*)allocB((size_t)256 * 147 * 2);  //    37,632
  __bf16* wb2  = (__bf16*)allocB((size_t)512 * 2304 * 2); // 1,179,648
  __bf16* wb3  = (__bf16*)allocB((size_t)1024 * 512 * 2); //   524,288
  __bf16* wb4  = (__bf16*)allocB((size_t)2048 * 1024 * 2);// 2,097,152
  __bf16* f1   = (__bf16*)allocB((size_t)BN * 256 * 56 * 56 * 2);
  __bf16* f2   = (__bf16*)allocB((size_t)BN * 512 * 28 * 28 * 2);
  __bf16* f3   = (__bf16*)allocB((size_t)BN * 1024 * 14 * 14 * 2);
  __bf16* f4   = (__bf16*)allocB((size_t)BN * 2048 * 49 * 2);   // persists
  __bf16* f4b  = (__bf16*)allocB((size_t)BN * 2048 * 49 * 2);   // branch
  __bf16* lin  = (__bf16*)allocB(NIMG * 2);
  float*  gvec = (float*)allocB((size_t)BN * 2048 * 4);
  float*  gf   = (float*)allocB((size_t)BN * 200 * 4);
  float*  cam  = (float*)allocB((size_t)BN * 49 * 4);
  float*  wscb = (float*)allocB((size_t)BN * IMGW * 4);
  float*  hscb = (float*)allocB((size_t)BN * IMGH * 4);
  float*  lfA  = (float*)allocB((size_t)BN * 2048 * 4);
  float*  lfB  = (float*)allocB((size_t)BN * 2048 * 4);
  int*    top  = (int*)allocB(BN * sizeof(int));
  int*    locs = (int*)allocB(BN * 4 * sizeof(int));

  // one-time fp32 -> bf16 conversions (weights + input image)
  cvt_bf16_kernel<<<cdiv((int)NIMG, 256), 256, 0, stream>>>(inputs, inbf, (int)NIMG);
  cvt_bf16_kernel<<<cdiv(256 * 147, 256), 256, 0, stream>>>(bw1, wb1, 256 * 147);
  cvt_bf16_kernel<<<cdiv(512 * 2304, 256), 256, 0, stream>>>(bw2, wb2, 512 * 2304);
  cvt_bf16_kernel<<<cdiv(1024 * 512, 256), 256, 0, stream>>>(bw3, wb3, 1024 * 512);
  cvt_bf16_kernel<<<cdiv(2048 * 1024, 256), 256, 0, stream>>>(bw4, wb4, 2048 * 1024);

  BBParams bb = { wb1, wb2, wb3, wb4, bb1, bb2, bb3, bb4 };

  float* out_sp  = (float*)d_out;          // (4,200)
  float* out_fam = out_sp + BN * 200;      // (4,38)
  float* out_ord = out_fam + BN * 38;      // (4,13)

  // main backbone + global pooled feature
  run_backbone(inbf, bb, f1, f2, f3, f4, stream);
  avgpool_kernel<<<cdiv(BN * 2048, 256), 256, 0, stream>>>(f4, gvec, BN * 2048, 49);

  struct Stage {
    const float *wc, *bc, *wc1, *bc1;
    const float* lfprev;
    float* lfout;
    float* fout;
    int On;
  } stages[3] = {
      { w_ord, b_ord, w_ord1, b_ord1, nullptr, lfA, out_ord, 13  },
      { w_fam, b_fam, w_fam1, b_fam1, lfA,     lfB, out_fam, 38  },
      { w_sp,  b_sp,  w_sp1,  b_sp1,  lfB,     lfA, out_sp,  200 },
  };

  for (int si = 0; si < 3; ++si) {
    const Stage& st = stages[si];
    int waves, thr;

    // gf = gvec @ wc^T + bc
    waves = BN * st.On; thr = waves * 32;
    linear_kernel<<<cdiv(thr, 128), 128, 0, stream>>>(gvec, st.wc, st.bc, gf,
                                                      BN, 2048, st.On);
    // top = argmax(gf)
    argmax_kernel<<<1, 32, 0, stream>>>(gf, top, BN, st.On);

    // CAM for top class only
    waves = BN * 49; thr = waves * 32;
    cam_kernel<<<cdiv(thr, 128), 128, 0, stream>>>(f4, st.wc, st.bc, top, cam,
                                                   BN, 2048, 49);
    // axis-max scores of upsampled CAM
    score_kernel<<<cdiv(BN * (IMGW + IMGH), 256), 256, 0, stream>>>(
        cam, wscb, hscb, BN, 7, 7, IMGH, IMGW);
    // interval search -> crop box
    objloc_kernel<<<1, 32, 0, stream>>>(wscb, hscb, locs, BN, IMGW, 0.5f);
    // crop-resize input (writes bf16 for the branch backbone)
    resize_kernel<<<cdiv(BN * 3 * IMGH * IMGW, 256), 256, 0, stream>>>(
        inputs, locs, lin, BN, 3, IMGH, IMGW);
    // branch backbone + pooled local feature
    run_backbone(lin, bb, f1, f2, f3, f4b, stream);
    avgpool_kernel<<<cdiv(BN * 2048, 256), 256, 0, stream>>>(f4b, st.lfout,
                                                             BN * 2048, 49);
    // (gf + ls) / 2
    waves = BN * st.On; thr = waves * 32;
    final_kernel<<<cdiv(thr, 128), 128, 0, stream>>>(
        gf, st.lfout, st.lfprev ? st.lfprev : st.lfout, st.wc1, st.bc1,
        st.fout, BN, 2048, st.On, st.lfprev != nullptr);
  }
}
